// ContrastMEMLoss_7052336300601
// MI455X (gfx1250) — compile-verified, matrix operand order checked by reading, not run on hardware
//
#include <hip/hip_runtime.h>

// ---------------------------------------------------------------------------
// ContrastMEMLoss for MI455X (gfx1250, wave32, WMMA + Tensor Data Mover)
//
// 3800x19000 logit matrix over D=256 (~37 GFLOP) fused with the masked
// log-softmax (fixed shift 1/TEMP=10 replaces the row max; the loss is
// shift-invariant up to the 1e-6 eps), so the logit matrix never touches
// memory. f16 operands / f32 accum via V_WMMA_F32_16X16X32_F16.
//
// Pass A streams the contrast matrix through LDS with TENSOR_LOAD_TO_LDS
// (TDM): two 64KB LDS panels are double-buffered; wave0 issues the DMA for
// step s+1 and waits s_wait_tensorcnt<=1 while all 8 waves consume panel s
// via ds_load_b128 -> v_wmma. This both overlaps copy with compute and
// fetches each B panel once per workgroup instead of once per wave.
// JAX threefry categorical sampling is not reproducible on-device; a
// deterministic ballot-scan anchor selection replaces it.
// ---------------------------------------------------------------------------

typedef __attribute__((ext_vector_type(16))) _Float16 v16h;
typedef __attribute__((ext_vector_type(8)))  _Float16 v8h;
typedef __attribute__((ext_vector_type(8)))  float    v8f;
typedef unsigned int u32x4 __attribute__((ext_vector_type(4)));
typedef int          i32x8 __attribute__((ext_vector_type(8)));
typedef int          i32x4 __attribute__((ext_vector_type(4)));

#define HW_     65536     // 256*256
#define B_      4
#define D_      256
#define CC_     19        // classes excluding ignore(0)
#define CACHE_  1000
#define NA_     50        // anchors per (b,c)
#define NROW_   3800      // NA*B*CC anchors
#define NROWP_  3808      // padded to 16
#define KCON_   19000     // CC*CACHE contrast rows
#define STEPC_  128       // contrast rows per TDM step (64KB panel)
#define KCONP_  19200     // padded to STEPC_ multiple (150 steps)
#define NSTEP_  (KCONP_ / STEPC_)
#define INVT_   10.0f     // 1/TEMP, also the fixed softmax shift
#define LOSSC_  (0.1f / 0.07f)   // TEMP/BASE_TEMP
#define WVN_    8         // waves per block in the GEMM kernels

// ---- deterministic anchor selection: first NA pixels with label==c && keep
__global__ __launch_bounds__(32) void k_select(const int* __restrict__ labels,
                                               const unsigned char* __restrict__ keep,
                                               int* __restrict__ sel)
{
  int pair = blockIdx.x;              // b*19 + (c-1)
  int b = pair / CC_;
  int c = pair % CC_ + 1;
  int lane = threadIdx.x;
  const int* lab = labels + b * HW_;
  const unsigned char* kp = keep + b * HW_;
  int count = 0;
  for (int base = 0; base < HW_ && count < NA_; base += 32) {
    int p = base + lane;
    bool m = (lab[p] == c) && (kp[p] != 0);
    unsigned long long bal = __ballot(m);
    int prefix = __popcll(bal & ((1ull << lane) - 1ull));
    int pos = count + prefix;
    if (m && pos < NA_) sel[pair * NA_ + pos] = p;
    count += __popcll(bal);
  }
  if (lane == 0) {
    int filled = count < NA_ ? count : NA_;
    int fill = (filled > 0) ? sel[pair * NA_] : 0;
    for (int i = filled; i < NA_; i++) sel[pair * NA_ + i] = fill;
  }
}

// ---- L2-normalize contrast rows (queue[1:] flattened) -> f16, pad with 0
__global__ __launch_bounds__(128) void k_norm_contrast(const float* __restrict__ queue,
                                                       _Float16* __restrict__ cn)
{
  int wave = threadIdx.x >> 5;
  int lane = threadIdx.x & 31;
  int row = blockIdx.x * 4 + wave;                 // 0..KCONP_-1
  _Float16* dst = cn + (size_t)row * D_;
  if (row >= KCON_) {
    for (int t = 0; t < 8; t++) dst[lane + 32 * t] = (_Float16)0.f;
    return;
  }
  const float* src = queue + (size_t)(CACHE_ + row) * D_;   // skip class 0
  float x[8]; float ss = 0.f;
  for (int t = 0; t < 8; t++) { x[t] = src[lane + 32 * t]; ss += x[t] * x[t]; }
  for (int off = 1; off < 32; off <<= 1) ss += __shfl_xor(ss, off, 32);
  float inv = 1.0f / fmaxf(sqrtf(ss), 1e-12f);
  for (int t = 0; t < 8; t++) dst[lane + 32 * t] = (_Float16)(x[t] * inv);
}

// ---- gather selected anchor pixels from feats [B,D,H,W], L2-normalize -> f16
__global__ __launch_bounds__(128) void k_gather_anchor(const float* __restrict__ feats,
                                                       const int* __restrict__ sel,
                                                       _Float16* __restrict__ an)
{
  int wave = threadIdx.x >> 5;
  int lane = threadIdx.x & 31;
  int n = blockIdx.x * 4 + wave;                   // 0..NROWP_-1
  _Float16* dst = an + (size_t)n * D_;
  if (n >= NROW_) {
    for (int t = 0; t < 8; t++) dst[lane + 32 * t] = (_Float16)0.f;
    return;
  }
  // row order matches reference: n = a*(B*CC) + b*CC + (c-1)
  int a = n / (B_ * CC_);
  int r = n % (B_ * CC_);                          // = b*CC + (c-1) = pair id
  int b = r / CC_;
  int p = sel[r * NA_ + a];
  const float* src = feats + (size_t)b * D_ * HW_ + p;
  float x[8]; float ss = 0.f;
  for (int t = 0; t < 8; t++) {
    int d = lane + 32 * t;
    x[t] = src[(size_t)d * HW_];
    ss += x[t] * x[t];
  }
  for (int off = 1; off < 32; off <<= 1) ss += __shfl_xor(ss, off, 32);
  float inv = 1.0f / fmaxf(sqrtf(ss), 1e-12f);
  for (int t = 0; t < 8; t++) dst[lane + 32 * t] = (_Float16)(x[t] * inv);
}

// ---- load a 16-half WMMA fragment per CDNA5 16-bit A/B layout:
// lane<16: k = {k0..k0+7, k0+16..k0+23}; lane>=16 handled by caller via k0+=8
__device__ __forceinline__ v16h load_frag(const _Float16* base, int k0)
{
  v8h lo = *(const v8h*)(base + k0);
  v8h hi = *(const v8h*)(base + k0 + 16);
  return __builtin_shufflevector(lo, hi, 0, 1, 2, 3, 4, 5, 6, 7,
                                 8, 9, 10, 11, 12, 13, 14, 15);
}

__device__ __forceinline__ void load_btile(v16h (&bf)[8], const _Float16* bb, int hi)
{
  #pragma unroll
  for (int kk = 0; kk < 8; kk++) bf[kk] = load_frag(bb, kk * 32 + 8 * hi);
}

__device__ __forceinline__ v8f wmma_tile(const v16h (&af)[8], const v16h (&bf)[8])
{
  v8f acc = {};
  #pragma unroll
  for (int kk = 0; kk < 8; kk++)
    acc = __builtin_amdgcn_wmma_f32_16x16x32_f16(false, af[kk], false, bf[kk],
                                                 (short)0, acc, false, false);
  return acc;
}

// ---- TDM: DMA a [STEPC_ x D_] f16 panel (64KB) of cn into LDS (2D descriptor)
__device__ __forceinline__ void tdm_load_panel(const _Float16* cn, int step,
                                               unsigned lds_byte_off)
{
  unsigned long long ga = (unsigned long long)(uintptr_t)cn +
                          (unsigned long long)step * STEPC_ * D_ * 2;
  u32x4 g0;
  g0[0] = 1u;                                            // count=1, user mode
  g0[1] = lds_byte_off;                                  // lds_addr
  g0[2] = (unsigned)ga;                                  // global_addr[31:0]
  g0[3] = (unsigned)((ga >> 32) & 0x1FFFFFFu) | (2u << 30); // addr[56:32]|type=2
  i32x8 g1;
  g1[0] = (1 << 16);                                     // data_size=2B, no flags
  g1[1] = (int)((unsigned)(D_ & 0xFFFF) << 16);          // tensor_dim0 lo16
  g1[2] = (int)(((unsigned)KCONP_ & 0xFFFFu) << 16);     // dim0 hi=0 | dim1 lo16
  g1[3] = (int)((unsigned)D_ << 16);                     // dim1 hi=0 | tile_dim0
  g1[4] = STEPC_;                                        // tile_dim1 | tile_dim2=0
  g1[5] = D_;                                            // tensor_dim0_stride lo32
  g1[6] = 0;                                             // stride hi | dim1_stride lo
  g1[7] = 0;
  i32x4 g2 = {0, 0, 0, 0};
  i32x4 g3 = {0, 0, 0, 0};
  i32x8 g4 = {0, 0, 0, 0, 0, 0, 0, 0};
  __builtin_amdgcn_tensor_load_to_lds(g0, g1, g2, g3, g4, 0);
}

// ---- pass A: neg_sum[row] = sum over negative columns of exp(logit - 10)
__device__ __forceinline__ void neg_epilogue(const v8f& acc, int col,
                                             const int (&rowcls)[8],
                                             float (&negacc)[8])
{
  bool colok = col < KCON_;
  int colcls = col / CACHE_ + 1;
  #pragma unroll
  for (int i = 0; i < 8; i++) {
    float l = acc[i] * INVT_ - INVT_;              // logit - fixed shift (10)
    float e = __expf(l);
    negacc[i] += (colok && (colcls != rowcls[i])) ? e : 0.f;
  }
}

__global__ __launch_bounds__(32 * WVN_) void k_neg(const _Float16* __restrict__ an,
                                                   const _Float16* __restrict__ cn,
                                                   float* __restrict__ negout)
{
  __shared__ _Float16 bbuf[2][STEPC_ * D_];        // 2 x 64KB LDS panels
  __shared__ float negsh[16];
  int lane = threadIdx.x & 31;
  int wave = threadIdx.x >> 5;
  if (threadIdx.x < 16) negsh[threadIdx.x] = 0.f;
  int row0 = blockIdx.x * 16;
  int hi = lane >> 4;
  int lm = lane & 15;

  const _Float16* ab = an + (size_t)(row0 + lm) * D_;
  v16h afrag[8];
  #pragma unroll
  for (int kk = 0; kk < 8; kk++) afrag[kk] = load_frag(ab, kk * 32 + 8 * hi);

  int rowcls[8];
  #pragma unroll
  for (int i = 0; i < 8; i++) {
    int rr = row0 + i + 8 * hi;                    // row for C element i
    rowcls[i] = (rr % (B_ * CC_)) % CC_ + 1;
  }
  float negacc[8];
  #pragma unroll
  for (int i = 0; i < 8; i++) negacc[i] = 0.f;

  unsigned lds0 = (unsigned)(uintptr_t)&bbuf[0][0];
  unsigned lds1 = (unsigned)(uintptr_t)&bbuf[1][0];
  if (wave == 0) tdm_load_panel(cn, 0, lds0);

  for (int s = 0; s < NSTEP_; s++) {
    int cur = s & 1;
    if (wave == 0) {
      if (s + 1 < NSTEP_) {
        tdm_load_panel(cn, s + 1, cur ? lds0 : lds1);
        __builtin_amdgcn_s_wait_tensorcnt(1);      // panel s done, s+1 in flight
      } else {
        __builtin_amdgcn_s_wait_tensorcnt(0);
      }
    }
    __syncthreads();                               // panel s visible to all waves
    {
      int cloc = wave * 16 + lm;                   // column slice for this wave
      const _Float16* bb = &bbuf[cur][cloc * D_];  // ds_load_b128 fragments
      v16h bf[8];
      load_btile(bf, bb, hi);
      v8f acc = wmma_tile(afrag, bf);
      neg_epilogue(acc, s * STEPC_ + cloc, rowcls, negacc);
    }
    __syncthreads();                               // done reading before overwrite
  }
  #pragma unroll
  for (int i = 0; i < 8; i++) {
    float v = negacc[i];
    for (int off = 1; off < 16; off <<= 1) v += __shfl_xor(v, off, 16);
    if (lm == 0) atomicAdd(&negsh[i + 8 * hi], v);
  }
  __syncthreads();
  if (threadIdx.x < 16) {
    int r = row0 + threadIdx.x;
    if (r < NROW_) negout[r] = negsh[threadIdx.x];
  }
}

// ---- pass B: recompute each row's positive block (its class's 1000 contrast
// rows), accumulate sum_pos[l - log(exp(l)+neg+1e-6)] -> scalar loss
__device__ __forceinline__ void pos_epilogue(const v8f& acc, int col,
                                             const float (&negv)[8],
                                             const bool (&rowok)[8],
                                             float (&posacc)[8])
{
  bool colok = col < CACHE_;
  #pragma unroll
  for (int i = 0; i < 8; i++) {
    float l = acc[i] * INVT_ - INVT_;
    float term = l - __logf(__expf(l) + negv[i]);
    posacc[i] += (colok && rowok[i]) ? term : 0.f;
  }
}

__global__ __launch_bounds__(32 * WVN_) void k_pos(const _Float16* __restrict__ an,
                                                   const _Float16* __restrict__ cn,
                                                   const float* __restrict__ negin,
                                                   float* __restrict__ out)
{
  __shared__ float possh[16];
  int lane = threadIdx.x & 31;
  int wave = threadIdx.x >> 5;
  if (threadIdx.x < 16) possh[threadIdx.x] = 0.f;
  __syncthreads();
  int cls = blockIdx.x / 13;            // 0..18 -> class cls+1
  int ttile = blockIdx.x % 13;          // 200 anchors/class -> 13 tiles of 16
  int tbase = ttile * 16;
  int hi = lane >> 4;
  int lm = lane & 15;

  // anchors of class cls+1 sit at rows 19*t + cls, t = 0..199
  int tA = tbase + lm; if (tA > 199) tA = 199;       // clamp pad rows
  const _Float16* ab = an + (size_t)(CC_ * tA + cls) * D_;
  v16h afrag[8];
  #pragma unroll
  for (int kk = 0; kk < 8; kk++) afrag[kk] = load_frag(ab, kk * 32 + 8 * hi);

  float negv[8]; bool rowok[8];
  #pragma unroll
  for (int i = 0; i < 8; i++) {
    int t = tbase + i + 8 * hi;
    rowok[i] = t < 200;
    int tt = rowok[i] ? t : 199;
    negv[i] = negin[CC_ * tt + cls] + 1e-6f;
  }
  float posacc[8];
  #pragma unroll
  for (int i = 0; i < 8; i++) posacc[i] = 0.f;

  const _Float16* cbase = cn + (size_t)cls * CACHE_ * D_;
  const int NT = 63;                                 // ceil(1000/16) col tiles
  for (int nt = wave; nt < NT; nt += WVN_) {
    int col = nt * 16 + lm;
    bool colok = col < CACHE_;
    int ccol = colok ? col : 0;
    v16h bf[8];
    load_btile(bf, cbase + (size_t)ccol * D_, hi);
    v8f acc = wmma_tile(afrag, bf);
    pos_epilogue(acc, col, negv, rowok, posacc);
  }
  #pragma unroll
  for (int i = 0; i < 8; i++) {
    float v = posacc[i];
    for (int off = 1; off < 16; off <<= 1) v += __shfl_xor(v, off, 16);
    if (lm == 0) atomicAdd(&possh[i + 8 * hi], v);
  }
  __syncthreads();
  if (threadIdx.x < 16) {
    int t = tbase + threadIdx.x;
    if (t < 200) {
      float mean_pos = possh[threadIdx.x] / (float)CACHE_;   // mask sum = 1000
      float contrib = -LOSSC_ * mean_pos / (float)NROW_;     // fold in mean()
      atomicAdd(out, contrib);
    }
  }
}

__global__ void k_zero(float* __restrict__ p, int n)
{
  int i = blockIdx.x * blockDim.x + threadIdx.x;
  if (i < n) p[i] = 0.f;
}

extern "C" void kernel_launch(void* const* d_in, const int* in_sizes, int n_in,
                              void* d_out, int out_size, void* d_ws, size_t ws_size,
                              hipStream_t stream)
{
  (void)in_sizes; (void)n_in; (void)ws_size;
  const float* feats  = (const float*)d_in[0];
  const float* output = (const float*)d_in[1];
  (void)output;  // entropy weights only bias the (non-reproducible) sampling
  const int* labels = (const int*)d_in[2];
  const unsigned char* keep = (const unsigned char*)d_in[3];
  const float* queue = (const float*)d_in[4];
  float* out = (float*)d_out;

  // workspace layout (~11.8 MB total)
  _Float16* cn = (_Float16*)d_ws;                      // 19200*256 f16
  _Float16* an = cn + (size_t)KCONP_ * D_;             //  3808*256 f16
  float* negbuf = (float*)(an + (size_t)NROWP_ * D_);  //  3808 f32
  int* sel = (int*)(negbuf + NROWP_);                  //  3800 i32

  k_zero<<<(out_size + 63) / 64, 64, 0, stream>>>(out, out_size);
  k_select<<<B_ * CC_, 32, 0, stream>>>(labels, keep, sel);
  k_norm_contrast<<<KCONP_ / 4, 128, 0, stream>>>(queue, cn);
  k_gather_anchor<<<NROWP_ / 4, 128, 0, stream>>>(feats, sel, an);
  k_neg<<<NROWP_ / 16, 32 * WVN_, 0, stream>>>(an, cn, negbuf);
  k_pos<<<CC_ * 13, 32 * WVN_, 0, stream>>>(an, cn, negbuf, out);
}
